// ExpressiveUpdateRule_50225347560012
// MI455X (gfx1250) — matrix operand order repaired and checked by reading, not compiled, not run on hardware
//
#include <hip/hip_runtime.h>

#define DIM 1024
#define TILE_M 16
#define LAMBDA_SQ 1024.0f

typedef __attribute__((ext_vector_type(16))) _Float16 v16h;
typedef __attribute__((ext_vector_type(8)))  float    v8f;

// ---------------------------------------------------------------------------
// Kernel 1: scale[b] = 1 / (lambda^2 + softplus(x[b]·eta_w + eta_b))
// ---------------------------------------------------------------------------
__global__ __launch_bounds__(256) void eta_scale_kernel(
    const float* __restrict__ x, const float* __restrict__ eta_w,
    const float* __restrict__ eta_b, float* __restrict__ scale_out)
{
    const int b   = blockIdx.x;
    const int tid = threadIdx.x;

    const float4 xv = *(const float4*)(x + (size_t)b * DIM + tid * 4);
    const float4 wv = *(const float4*)(eta_w + tid * 4);
    float s = xv.x * wv.x + xv.y * wv.y + xv.z * wv.z + xv.w * wv.w;

    #pragma unroll
    for (int off = 16; off >= 1; off >>= 1) s += __shfl_xor(s, off, 32);

    __shared__ float red[8];
    if ((tid & 31) == 0) red[tid >> 5] = s;
    __syncthreads();

    if (tid == 0) {
        float t = 0.0f;
        #pragma unroll
        for (int i = 0; i < 8; ++i) t += red[i];
        float z  = t + eta_b[0];
        float sp = (z > 20.0f) ? z : log1pf(expf(z));   // softplus
        scale_out[b] = 1.0f / (LAMBDA_SQ + sp);
    }
}

// ---------------------------------------------------------------------------
// Kernel 2: fused GEMV (WMMA f16->f32) + rank-1 update, 16 rows per block.
//   Phase 1: y_hat[m] = sum_k W[b,row0+m,k] * x[b,k]  via v_wmma_f32_16x16x32_f16
//   Phase 2: out = W + coeff[m]*x,  coeff[m] = -scale*(y_hat[m]-grad[m])
// ---------------------------------------------------------------------------
__global__ __launch_bounds__(256) void sm_delta_kernel(
    const float* __restrict__ W, const float* __restrict__ x,
    const float* __restrict__ grad, const float* __restrict__ scale_arr,
    float* __restrict__ out)
{
    const int b    = blockIdx.y;
    const int row0 = blockIdx.x * TILE_M;
    const int tid  = threadIdx.x;
    const int lane = tid & 31;
    const int wave = tid >> 5;

    __shared__ __align__(16) float xsh[DIM];   // 4 KB: x[b] row
    __shared__ float yhat[TILE_M];
    __shared__ float gsh[TILE_M];

    // cache x[b] in LDS; init reduction buffer; pull grad tile
    {
        float4 v = *(const float4*)(x + (size_t)b * DIM + tid * 4);
        *(float4*)(xsh + tid * 4) = v;
    }
    if (tid < TILE_M) {
        yhat[tid] = 0.0f;
        gsh[tid]  = grad[(size_t)b * DIM + row0 + tid];
    }
    __syncthreads();

    const float* Wtile = W + ((size_t)b * DIM + row0) * DIM;

    // ---- Phase 1: each wave covers a K-chunk of 128 (4 WMMAs, C chained) ----
    const int m  = lane & 15;      // A-matrix row held by this lane
    const int hi = lane >> 4;      // lane half selects K sub-range (+8 for A and B)
    v8f c = {};
    const int kbase = wave * 128;
    const float* wrow = Wtile + (size_t)m * DIM + hi * 8;

    #pragma unroll
    for (int kk = 0; kk < 4; ++kk) {
        const int k0 = kbase + kk * 32;

        // A: 16-bit 16x32 layout — lane<16: VGPR j holds K={2j,2j+1} (+16 for j>=4);
        //    lane>=16: same rows, K offset +8.
        v16h a;
        #pragma unroll
        for (int j = 0; j < 8; ++j) {
            const int koff = (j & 3) * 2 + (j >> 2) * 16;
            float2 w2 = *(const float2*)(wrow + k0 + koff);
            a[2 * j]     = (_Float16)w2.x;
            a[2 * j + 1] = (_Float16)w2.y;
        }

    // B: 32x16 layout — same K order as A: lane<16 holds K=0..7,16..23; lane>=16 K=8..15,24..31,
        //    2 K-values per VGPR; columns replicated with x so every N col = y_hat.
        v16h bv;
        const float* xp = xsh + k0 + hi * 8;
        #pragma unroll
        for (int e = 0; e < 16; e += 2) {
            float2 x2 = *(const float2*)(xp + e + (e >> 3) * 8);
            bv[e]     = (_Float16)x2.x;
            bv[e + 1] = (_Float16)x2.y;
        }

        c = __builtin_amdgcn_wmma_f32_16x16x32_f16(
                /*neg_a=*/false, a, /*neg_b=*/false, bv,
                /*c_mod=*/(short)0, c, /*reuse_a=*/false, /*reuse_b=*/false);
    }

    // C/D layout: VGPR r, lanes 0-15 -> M=r; lanes 16-31 -> M=8+r (all N equal).
    if (lane == 0) {
        #pragma unroll
        for (int r = 0; r < 8; ++r) atomicAdd(&yhat[r], c[r]);
    } else if (lane == 16) {
        #pragma unroll
        for (int r = 0; r < 8; ++r) atomicAdd(&yhat[8 + r], c[r]);
    }
    __syncthreads();

    if (tid < TILE_M) {
        const float scale = scale_arr[b];
        yhat[tid] = -scale * (yhat[tid] - gsh[tid]);   // coeff[m]
    }
    __syncthreads();

    // ---- Phase 2: stream tile back out with rank-1 update (L0/L2-hot re-read) ----
    const float4 x4  = *(const float4*)(xsh + tid * 4);
    const float4* Wp = (const float4*)Wtile;
    float4* Op = (float4*)(out + ((size_t)b * DIM + row0) * DIM);

    #pragma unroll
    for (int mm = 0; mm < TILE_M; ++mm) {
        const float coeff = yhat[mm];
        float4 w4 = Wp[mm * (DIM / 4) + tid];
        float4 o;
        o.x = fmaf(coeff, x4.x, w4.x);
        o.y = fmaf(coeff, x4.y, w4.y);
        o.z = fmaf(coeff, x4.z, w4.z);
        o.w = fmaf(coeff, x4.w, w4.w);
        Op[mm * (DIM / 4) + tid] = o;
    }
}

// ---------------------------------------------------------------------------
extern "C" void kernel_launch(void* const* d_in, const int* in_sizes, int n_in,
                              void* d_out, int out_size, void* d_ws, size_t ws_size,
                              hipStream_t stream) {
    const float* W    = (const float*)d_in[0];   // [B, D, D]
    const float* x    = (const float*)d_in[1];   // [B, D]
    const float* grad = (const float*)d_in[2];   // [B, D]
    const float* ew   = (const float*)d_in[3];   // [1, D]
    const float* eb   = (const float*)d_in[4];   // [1]
    float* out   = (float*)d_out;                // [B, D, D]
    float* scale = (float*)d_ws;                 // [B] scratch

    const int B = in_sizes[1] / DIM;             // 128

    eta_scale_kernel<<<B, 256, 0, stream>>>(x, ew, eb, scale);
    sm_delta_kernel<<<dim3(DIM / TILE_M, B), 256, 0, stream>>>(W, x, grad, scale, out);
}
